// GLR_SSM_EAD_Memory_5136780886732
// MI455X (gfx1250) — matrix operand order repaired and checked by reading, not compiled, not run on hardware
//
#include <hip/hip_runtime.h>
#include <hip/hip_bf16.h>
#include <math.h>

typedef __bf16 bf16;
typedef __attribute__((ext_vector_type(16))) __bf16 v16bf;
typedef __attribute__((ext_vector_type(8)))  __bf16 v8bf;
typedef __attribute__((ext_vector_type(4)))  __bf16 v4bf;
typedef __attribute__((ext_vector_type(8)))  float  v8f;
typedef __attribute__((ext_vector_type(4)))  unsigned int v4u;
typedef __attribute__((ext_vector_type(4)))  int v4i;
typedef __attribute__((ext_vector_type(8)))  int v8i;

#define BB   8
#define TT   4096
#define DD   768
#define RR   16
#define DBK  256
#define BT   (BB*TT)      /* 32768 rows */
#define NCH  32
#define CL   (TT/NCH)     /* 128 */
#define OUT_ROWS 128      /* rows per k_out block (8 waves x 16) */

#if __has_builtin(__builtin_amdgcn_tensor_load_to_lds)
#define HAVE_TDM 1
#endif
#if __has_include(<hip/amd_detail/amd_gfx1250_TDM.h>)
#define TDM_ARITY6 1
#endif

union Frag16 { v16bf v; v8bf h[2]; uint4 q[2]; };

// ---------------------------------------------------------------------------
// WMMA fragment loaders (bf16, 16x16x32), A and B share one contiguous K order.
// A (16x32, MxK): lane m = lane&15, hi = lane>>4; K = hi*16 + e (16 contiguous)
// => two b128 loads.  B (32x16, KxN) from row-major W (N x K): lane n, K =
// hi*16 + e => 16 contiguous bf16 = two b128 loads (global or LDS).
// ---------------------------------------------------------------------------
__device__ __forceinline__ v16bf load_a_bf16(const bf16* __restrict__ X, int ld,
                                             int m0, int k0, int lane) {
  int m = lane & 15, hi = lane >> 4;
  const bf16* p = X + (size_t)(m0 + m) * ld + k0 + hi * 16;
  Frag16 f;
  f.q[0] = *(const uint4*)(p);
  f.q[1] = *(const uint4*)(p + 8);
  return f.v;
}

__device__ __forceinline__ v16bf load_b_bf16(const bf16* __restrict__ W, int ld,
                                             int n0, int k0, int lane) {
  int n = lane & 15, hi = lane >> 4;
  const bf16* p = W + (size_t)(n0 + n) * ld + k0 + hi * 16;
  Frag16 f;
  f.q[0] = *(const uint4*)(p);
  f.q[1] = *(const uint4*)(p + 8);
  return f.v;
}

__device__ __forceinline__ v8f wmma_bf16(v16bf a, v16bf b, v8f c) {
  return __builtin_amdgcn_wmma_f32_16x16x32_bf16(false, a, false, b,
                                                 (short)0, c, false, false);
}

__device__ __forceinline__ float sigmoidf_(float x) {
  return 1.0f / (1.0f + __expf(-x));
}
__device__ __forceinline__ float gelu_exact(float x) {
  return 0.5f * x * (1.0f + erff(x * 0.70710678118654752f));
}

// ---------------------------------------------------------------------------
// TDM: 1-D contiguous tensor_load_to_lds (nelem bf16 elements).  D# fields per
// CDNA5 ISA sec 8.3/8.4: group0 = {count=1, lds_addr, global_addr, type=2},
// group1 = {data_size=2B, tensor_dim0 = tile_dim0 = nelem, tensor_dim1=1}.
// ---------------------------------------------------------------------------
#if defined(HAVE_TDM)
__device__ __forceinline__ void tdm_load_1d(unsigned lds_off, const void* gptr,
                                            unsigned nelem) {
  unsigned long long ga = (unsigned long long)(uintptr_t)gptr;
  v4u g0;
  g0[0] = 1u;                                            // count=1, user D#
  g0[1] = lds_off;                                       // LDS byte address
  g0[2] = (unsigned)(ga & 0xFFFFFFFFu);                  // global_addr[31:0]
  g0[3] = (unsigned)((ga >> 32) & 0x01FFFFFFu) | (2u << 30); // [56:32] | type=2
  v8i g1;
  g1[0] = (int)(1u << 16);                               // data_size = 2 bytes
  g1[1] = (int)((nelem & 0xFFFFu) << 16);                // tensor_dim0[15:0]
  g1[2] = (int)(((nelem >> 16) & 0xFFFFu) | (1u << 16)); // td0[31:16] | td1=1
  g1[3] = (int)((nelem & 0xFFFFu) << 16);                // tile_dim0 (<=65535)
  g1[4] = 0;                                             // tile_dim1/2 unused
  g1[5] = (int)nelem;                                    // dim0_stride[31:0]
  g1[6] = 0;
  g1[7] = 0;
  v4i z4 = {0, 0, 0, 0};
#if defined(TDM_ARITY6)
  v8i z8 = {0, 0, 0, 0, 0, 0, 0, 0};
  __builtin_amdgcn_tensor_load_to_lds(g0, g1, z4, z4, z8, 0);
#else
  __builtin_amdgcn_tensor_load_to_lds(g0, g1, z4, z4, 0);
#endif
}
#endif

// ---------------------------------------------------------------------------
// fp32 -> bf16 convert (vectorized float4 -> 4x bf16)
// ---------------------------------------------------------------------------
__global__ void k_cvt_bf16(const float* __restrict__ s, bf16* __restrict__ d, int n4) {
  int i = blockIdx.x * blockDim.x + threadIdx.x;
  if (i < n4) {
    float4 v = ((const float4*)s)[i];
    v4bf o;
    o[0] = (bf16)v.x; o[1] = (bf16)v.y; o[2] = (bf16)v.z; o[3] = (bf16)v.w;
    ((v4bf*)d)[i] = o;
  }
}

// ---------------------------------------------------------------------------
// Low-rank path: Z = x V^T (16x16), Bu = Z U^T (16x768). One wave / 16 rows.
// Z staged through LDS (K padded 16->32 with zeros) to become an A operand.
// ---------------------------------------------------------------------------
__global__ void __launch_bounds__(32) k_zbu(const bf16* __restrict__ xb,
                                            const bf16* __restrict__ Vb,
                                            const bf16* __restrict__ Ub,
                                            bf16* __restrict__ Bub) {
  int lane = threadIdx.x;
  int row0 = blockIdx.x * 16;
  int n = lane & 15, hi = lane >> 4;

  v8f z = {0.f,0.f,0.f,0.f,0.f,0.f,0.f,0.f};
  for (int k0 = 0; k0 < DD; k0 += 32) {
    v16bf a = load_a_bf16(xb, DD, row0, k0, lane);
    v16bf b = load_b_bf16(Vb, DD, 0, k0, lane);
    z = wmma_bf16(a, b, z);
  }

  __shared__ __align__(16) bf16 zl[16 * 32];
#pragma unroll
  for (int j = 0; j < 8; ++j) {
    int r = j + 8 * hi;
    zl[r * 32 + n]      = (bf16)z[j];
    zl[r * 32 + 16 + n] = (bf16)0.0f;   // K padding 16..31
  }
  __syncthreads();

  v16bf az = load_a_bf16(zl, 32, 0, 0, lane);

  for (int n0 = 0; n0 < DD; n0 += 16) {
    Frag16 fb;
    fb.q[0] = make_uint4(0u, 0u, 0u, 0u);
    fb.q[1] = make_uint4(0u, 0u, 0u, 0u);
    if (hi == 0) {                       // K 0..15 real (U_w row), K 16..31 zero
      const bf16* p = Ub + (size_t)(n0 + n) * RR;
      fb.q[0] = *(const uint4*)(p);
      fb.q[1] = *(const uint4*)(p + 8);
    }
    v8f acc = {0.f,0.f,0.f,0.f,0.f,0.f,0.f,0.f};
    acc = wmma_bf16(az, fb.v, acc);
#pragma unroll
    for (int j = 0; j < 8; ++j) {
      int r = row0 + j + 8 * hi;
      Bub[(size_t)r * DD + n0 + n] = (bf16)acc[j];
    }
  }
}

// ---------------------------------------------------------------------------
// Evidence path: h = gelu(x E1^T + b1) (16x256 via WMMA), alpha = sigmoid(h E2^T + b2).
// alpha partials reduced across the 16 N-lanes of each half-wave via shfl_xor.
// ---------------------------------------------------------------------------
__global__ void __launch_bounds__(32) k_evidence(const bf16* __restrict__ xb,
                                                 const bf16* __restrict__ E1b,
                                                 const float* __restrict__ E1bias,
                                                 const float* __restrict__ E2w,
                                                 const float* __restrict__ E2b,
                                                 float* __restrict__ alpha) {
  int lane = threadIdx.x;
  int row0 = blockIdx.x * 16;
  int n = lane & 15, hi = lane >> 4;

  float p0[8], p1[8], p2[8];
#pragma unroll
  for (int j = 0; j < 8; ++j) { p0[j] = 0.f; p1[j] = 0.f; p2[j] = 0.f; }

  for (int t = 0; t < DBK / 16; ++t) {
    v8f acc = {0.f,0.f,0.f,0.f,0.f,0.f,0.f,0.f};
    for (int k0 = 0; k0 < DD; k0 += 32) {
      v16bf a = load_a_bf16(xb, DD, row0, k0, lane);
      v16bf b = load_b_bf16(E1b, DD, t * 16, k0, lane);
      acc = wmma_bf16(a, b, acc);
    }
    int col = t * 16 + n;
    float bias = E1bias[col];
    float w0 = E2w[0 * DBK + col], w1 = E2w[1 * DBK + col], w2 = E2w[2 * DBK + col];
#pragma unroll
    for (int j = 0; j < 8; ++j) {
      float hv = gelu_exact(acc[j] + bias);
      p0[j] += hv * w0; p1[j] += hv * w1; p2[j] += hv * w2;
    }
  }
#pragma unroll
  for (int m = 1; m < 16; m <<= 1) {
#pragma unroll
    for (int j = 0; j < 8; ++j) {
      p0[j] += __shfl_xor(p0[j], m, 32);
      p1[j] += __shfl_xor(p1[j], m, 32);
      p2[j] += __shfl_xor(p2[j], m, 32);
    }
  }
  if (n == 0) {
    float b0 = E2b[0], b1 = E2b[1], b2 = E2b[2];
#pragma unroll
    for (int j = 0; j < 8; ++j) {
      size_t r = (size_t)(row0 + j + 8 * hi);
      alpha[r * 3 + 0] = sigmoidf_(p0[j] + b0);
      alpha[r * 3 + 1] = sigmoidf_(p1[j] + b1);
      alpha[r * 3 + 2] = sigmoidf_(p2[j] + b2);
    }
  }
}

// ---------------------------------------------------------------------------
// Chunked parallel scan: s_t = lam_t s_{t-1} + Bu_t ; lam = sigmoid(a_eff)
// (identity: exp(-softplus(-a)) == sigmoid(a)).
// ---------------------------------------------------------------------------
__global__ void k_scan1(const bf16* __restrict__ Bub, const float* __restrict__ alpha,
                        const float* __restrict__ a_base, const float* __restrict__ a_delta,
                        float* __restrict__ Ak, float* __restrict__ bk) {
  int d = blockIdx.x * blockDim.x + threadIdx.x;
  int chunk = blockIdx.y, b = blockIdx.z;
  float tnh = tanhf(a_delta[d]);
  float ab  = a_base[d];
  int c = d >> 8;
  float s = 0.f, A = 1.f;
  int t0 = chunk * CL;
  for (int t = t0; t < t0 + CL; ++t) {
    size_t rt = (size_t)(b * TT + t);
    float lam = sigmoidf_(ab + alpha[rt * 3 + c] * tnh);
    float bu  = (float)Bub[rt * DD + d];
    s = lam * s + bu;
    A *= lam;
  }
  size_t idx = (size_t)(b * DD + d) * NCH + chunk;
  Ak[idx] = A; bk[idx] = s;
}

__global__ void k_scan2(const float* __restrict__ Ak, const float* __restrict__ bk,
                        float* __restrict__ carry) {
  int idx = blockIdx.x * blockDim.x + threadIdx.x;
  if (idx >= BB * DD) return;
  float s = 0.f;
  for (int k = 0; k < NCH; ++k) {
    size_t i = (size_t)idx * NCH + k;
    carry[i] = s;
    s = Ak[i] * s + bk[i];
  }
}

__global__ void k_scan3(const bf16* __restrict__ Bub, const float* __restrict__ alpha,
                        const float* __restrict__ a_base, const float* __restrict__ a_delta,
                        const float* __restrict__ carry, bf16* __restrict__ Sb) {
  int d = blockIdx.x * blockDim.x + threadIdx.x;
  int chunk = blockIdx.y, b = blockIdx.z;
  float tnh = tanhf(a_delta[d]);
  float ab  = a_base[d];
  int c = d >> 8;
  float s = carry[(size_t)(b * DD + d) * NCH + chunk];
  int t0 = chunk * CL;
  for (int t = t0; t < t0 + CL; ++t) {
    size_t rt = (size_t)(b * TT + t);
    float lam = sigmoidf_(ab + alpha[rt * 3 + c] * tnh);
    float bu  = (float)Bub[rt * DD + d];
    s = lam * s + bu;
    Sb[rt * DD + d] = (bf16)s;
  }
}

// ---------------------------------------------------------------------------
// Fused output: y = S C^T, g = sigmoid(x G^T + G_b), out = x + gelu(y)*g.
// 8-wave blocks (128 rows x 16 cols).  Wave 0 DMAs the 16x768 C/G weight
// panels (contiguous 24 KB each) into LDS via the Tensor Data Mover, waits on
// TENSORcnt, then all 8 waves consume B fragments from LDS (ds_load_b128),
// amortizing the weight traffic 8x.  Residual/out use non-temporal hints so
// the streamed 200 MB fp32 doesn't evict L2-resident bf16 operands.
// ---------------------------------------------------------------------------
__global__ void __launch_bounds__(256) k_out(const bf16* __restrict__ Sb,
                                             const bf16* __restrict__ xb,
                                             const bf16* __restrict__ Cb,
                                             const bf16* __restrict__ Gb,
                                             const float* __restrict__ x,
                                             const float* __restrict__ Gbias,
                                             float* __restrict__ out) {
  __shared__ __align__(16) bf16 cpan[16 * DD];
  __shared__ __align__(16) bf16 gpan[16 * DD];
  int tid  = threadIdx.x;
  int lane = tid & 31;
  int wave = tid >> 5;
  int nbase = blockIdx.y * 16;
  int row0  = blockIdx.x * OUT_ROWS + wave * 16;
  int n = lane & 15, hi = lane >> 4;

#if defined(HAVE_TDM)
  if (wave == 0) {
    tdm_load_1d((unsigned)(uintptr_t)cpan, Cb + (size_t)nbase * DD, 16 * DD);
    tdm_load_1d((unsigned)(uintptr_t)gpan, Gb + (size_t)nbase * DD, 16 * DD);
#if __has_builtin(__builtin_amdgcn_s_wait_tensorcnt)
    __builtin_amdgcn_s_wait_tensorcnt(0);
#else
    asm volatile("s_wait_tensorcnt 0x0" ::: "memory");
#endif
  }
#else
  for (int i = tid; i < 16 * DD / 8; i += 256) {
    ((uint4*)cpan)[i] = ((const uint4*)(Cb + (size_t)nbase * DD))[i];
    ((uint4*)gpan)[i] = ((const uint4*)(Gb + (size_t)nbase * DD))[i];
  }
#endif
  __syncthreads();

  v8f accY = {0.f,0.f,0.f,0.f,0.f,0.f,0.f,0.f};
  v8f accG = {0.f,0.f,0.f,0.f,0.f,0.f,0.f,0.f};

  for (int k0 = 0; k0 < DD; k0 += 32) {
    v16bf aS = load_a_bf16(Sb, DD, row0, k0, lane);
    v16bf aX = load_a_bf16(xb, DD, row0, k0, lane);
    if (k0 + 32 < DD) {  // global_prefetch_b8 of next K panel of A operands
      __builtin_prefetch(Sb + (size_t)(row0 + n) * DD + k0 + 32, 0, 1);
      __builtin_prefetch(xb + (size_t)(row0 + n) * DD + k0 + 32, 0, 1);
    }
    v16bf bC = load_b_bf16(cpan, DD, 0, k0, lane);
    v16bf bG = load_b_bf16(gpan, DD, 0, k0, lane);
    accY = wmma_bf16(aS, bC, accY);
    accG = wmma_bf16(aX, bG, accG);
  }

  int col = nbase + n;
  float gb = Gbias[col];
#pragma unroll
  for (int j = 0; j < 8; ++j) {
    size_t r = (size_t)(row0 + j + 8 * hi);
    float y  = accY[j];
    float g  = sigmoidf_(accG[j] + gb);
    float xv = __builtin_nontemporal_load(&x[r * DD + col]);
    __builtin_nontemporal_store(xv + gelu_exact(y) * g, &out[r * DD + col]);
  }
}

// ---------------------------------------------------------------------------
extern "C" void kernel_launch(void* const* d_in, const int* in_sizes, int n_in,
                              void* d_out, int out_size, void* d_ws, size_t ws_size,
                              hipStream_t stream) {
  const float* x       = (const float*)d_in[0];
  const float* Vw      = (const float*)d_in[1];
  const float* Uw      = (const float*)d_in[2];
  const float* E1w     = (const float*)d_in[3];
  const float* E1bias  = (const float*)d_in[4];
  const float* E2w     = (const float*)d_in[5];
  const float* E2bias  = (const float*)d_in[6];
  const float* a_base  = (const float*)d_in[7];
  const float* a_delta = (const float*)d_in[8];
  const float* Cw      = (const float*)d_in[9];
  const float* Gw      = (const float*)d_in[10];
  const float* Gbias   = (const float*)d_in[11];
  float* out = (float*)d_out;

  char* ws = (char*)d_ws;
  size_t off = 0;
  auto alloc = [&](size_t bytes) -> char* {
    char* p = ws + off;
    off = (off + bytes + 255) & ~(size_t)255;
    return p;
  };
  bf16*  xb    = (bf16*)alloc((size_t)BT * DD * 2);
  bf16*  Bub   = (bf16*)alloc((size_t)BT * DD * 2);
  bf16*  Sb    = (bf16*)alloc((size_t)BT * DD * 2);
  float* alpha = (float*)alloc((size_t)BT * 3 * 4);
  float* Ak    = (float*)alloc((size_t)BB * DD * NCH * 4);
  float* bk    = (float*)alloc((size_t)BB * DD * NCH * 4);
  float* carry = (float*)alloc((size_t)BB * DD * NCH * 4);
  bf16*  Vb    = (bf16*)alloc((size_t)RR * DD * 2);
  bf16*  Ub    = (bf16*)alloc((size_t)DD * RR * 2);
  bf16*  E1b   = (bf16*)alloc((size_t)DBK * DD * 2);
  bf16*  Cb    = (bf16*)alloc((size_t)DD * DD * 2);
  bf16*  Gb    = (bf16*)alloc((size_t)DD * DD * 2);

  auto cvt = [&](const float* s, bf16* d, int nelem) {
    int n4 = nelem / 4;
    k_cvt_bf16<<<(n4 + 255) / 256, 256, 0, stream>>>(s, d, n4);
  };
  cvt(x,   xb,  BT * DD);
  cvt(Vw,  Vb,  RR * DD);
  cvt(Uw,  Ub,  DD * RR);
  cvt(E1w, E1b, DBK * DD);
  cvt(Cw,  Cb,  DD * DD);
  cvt(Gw,  Gb,  DD * DD);

  k_zbu<<<BT / 16, 32, 0, stream>>>(xb, Vb, Ub, Bub);
  k_evidence<<<BT / 16, 32, 0, stream>>>(xb, E1b, E1bias, E2w, E2bias, alpha);
  k_scan1<<<dim3(DD / 256, NCH, BB), 256, 0, stream>>>(Bub, alpha, a_base, a_delta, Ak, bk);
  k_scan2<<<(BB * DD + 255) / 256, 256, 0, stream>>>(Ak, bk, carry);
  k_scan3<<<dim3(DD / 256, NCH, BB), 256, 0, stream>>>(Bub, alpha, a_base, a_delta, carry, Sb);
  k_out<<<dim3(BT / OUT_ROWS, DD / 16), 256, 0, stream>>>(Sb, xb, Cb, Gb, x, Gbias, out);
}
